// MambaVisionMixerBlock_80917183857331
// MI455X (gfx1250) — matrix operand order repaired and checked by reading, not compiled, not run on hardware
//
#include <hip/hip_runtime.h>
#include <hip/hip_bf16.h>

// ---------------------------------------------------------------------------
// MambaVision mixer block for gfx1250 (MI455X), wave32 + WMMA bf16 path.
// B=2, L=4096, D=1024, DS=16, RK=64, conv K=3.
// 4 fused GEMMs (v_wmma_f32_16x16x32_bf16), 128x128 block tile, 8 waves,
// async-to-LDS double-buffered pipeline (ASYNCcnt) with register fallback.
// ---------------------------------------------------------------------------

typedef __attribute__((ext_vector_type(16))) __bf16 v16bf;
typedef __attribute__((ext_vector_type(8)))  float  v8f;

union FragBF { uint4 q[2]; v16bf v; };

#if defined(__gfx1250__) && \
    __has_builtin(__builtin_amdgcn_global_load_async_to_lds_b128) && \
    __has_builtin(__builtin_amdgcn_s_wait_asynccnt)
#define USE_ASYNC 1
typedef int v4i_gcc __attribute__((vector_size(16)));
using as1_v4i = __attribute__((address_space(1))) v4i_gcc;
using as3_v4i = __attribute__((address_space(3))) v4i_gcc;
#else
#define USE_ASYNC 0
#endif

__device__ __forceinline__ unsigned short f2bf(float f) {
    unsigned int u = __float_as_uint(f);
    unsigned int r = u + 0x7FFFu + ((u >> 16) & 1u);
    return (unsigned short)(r >> 16);
}

// ---------------------------------------------------------------------------
// Place fp32 [K][N] (row-major) transposed into bf16 dst at
// dst[(noff+n)*Kld + koff+k].  (Bt layout: one output row per N, contiguous K.)
// ---------------------------------------------------------------------------
__global__ void k_place_bf16(const float* __restrict__ src,
                             unsigned short* __restrict__ dst,
                             int K, int N, int Kld, int koff, int noff) {
    int idx = blockIdx.x * blockDim.x + threadIdx.x;
    if (idx >= N * K) return;
    int n = idx / K;
    int k = idx - n * K;
    dst[(size_t)(noff + n) * Kld + koff + k] = f2bf(src[(size_t)k * N + n]);
}

__global__ void k_zero_u16(unsigned short* __restrict__ p, int n) {
    int idx = blockIdx.x * blockDim.x + threadIdx.x;
    if (idx < n) p[idx] = 0;
}

// ---------------------------------------------------------------------------
// LayerNorm over D, fp32 in -> bf16 out.  One block per row.
// ---------------------------------------------------------------------------
__global__ void k_layernorm(const float* __restrict__ x,
                            const float* __restrict__ gamma,
                            const float* __restrict__ beta,
                            unsigned short* __restrict__ out, int D) {
    __shared__ float s1[256], s2[256];
    int row = blockIdx.x, tid = threadIdx.x;
    const float* xr = x + (size_t)row * D;
    float sum = 0.f, sq = 0.f;
    for (int d = tid; d < D; d += 256) { float v = xr[d]; sum += v; sq += v * v; }
    s1[tid] = sum; s2[tid] = sq; __syncthreads();
    for (int s = 128; s > 0; s >>= 1) {
        if (tid < s) { s1[tid] += s1[tid + s]; s2[tid] += s2[tid + s]; }
        __syncthreads();
    }
    float mu  = s1[0] / D;
    float var = s2[0] / D - mu * mu;
    float inv = rsqrtf(var + 1e-5f);
    for (int d = tid; d < D; d += 256) {
        float v = (xr[d] - mu) * inv * gamma[d] + beta[d];
        out[(size_t)row * D + d] = f2bf(v);
    }
}

// ---------------------------------------------------------------------------
// WMMA bf16 GEMM:  C[M][ldc] = A[M][ldA(:K)] * Bt[Npad][K]^T  (+ epilogue)
//   A  : bf16 [M][ldA], M % 128 == 0, K % 32 == 0, ldA % 8 == 0
//   Bt : bf16 [Npad][K], Npad % 128 == 0 (rows >= N zero-filled)
// 256 threads = 8 waves; block tile 128x128; wave tile 32x64 (2x4 WMMA).
// Double-buffered LDS; staging via async-to-LDS (ASYNCcnt) when available.
// EPI: 0 = f32 store, 1 = bf16 store, 2 = f32 + residual[row*ldc+col].
// ---------------------------------------------------------------------------
template <int EPI>
__global__ __launch_bounds__(256)
void k_gemm_wmma(const unsigned short* __restrict__ A,
                 const unsigned short* __restrict__ Bt,
                 void* __restrict__ Cout,
                 const float* __restrict__ residual,
                 int M, int N, int K, int ldA, int ldc) {
    __shared__ __align__(16) unsigned short lsA[2][128 * 40];  // 80B rows
    __shared__ __align__(16) unsigned short lsB[2][128 * 40];

    const int tid    = threadIdx.x;
    const int lane   = tid & 31;
    const int wave   = tid >> 5;          // 0..7
    const int halfId = lane >> 4;
    const int ln     = lane & 15;
    const int wm     = wave >> 1;         // 0..3  (m quadrant, 32 rows each)
    const int wn     = wave & 1;          // 0..1  (n quadrant, 64 cols each)
    const int blockM = blockIdx.x * 128;
    const int blockN = blockIdx.y * 128;

    // Staging: tile = 128 rows x 32 k = 512 x 16B chunks; 2 chunks/thread/mat.
    const int c0 = tid, c1 = tid + 256;
    const int r0 = c0 >> 2, o0 = (c0 & 3) * 8;   // in bf16 elements
    const int r1 = c1 >> 2, o1 = (c1 & 3) * 8;
    const unsigned short* pa0 = A  + (size_t)(blockM + r0) * ldA + o0;
    const unsigned short* pa1 = A  + (size_t)(blockM + r1) * ldA + o1;
    const unsigned short* pb0 = Bt + (size_t)(blockN + r0) * K + o0;
    const unsigned short* pb1 = Bt + (size_t)(blockN + r1) * K + o1;
    unsigned short* sA0 = &lsA[0][r0 * 40 + o0];
    unsigned short* sA1 = &lsA[0][r1 * 40 + o1];
    unsigned short* sB0 = &lsB[0][r0 * 40 + o0];
    unsigned short* sB1 = &lsB[0][r1 * 40 + o1];
    const int bufStride = 128 * 40;

    v8f acc[2][4];
    const v8f vzero = {0.f, 0.f, 0.f, 0.f, 0.f, 0.f, 0.f, 0.f};
    #pragma unroll
    for (int i = 0; i < 2; ++i)
        #pragma unroll
        for (int j = 0; j < 4; ++j) acc[i][j] = vzero;

#if USE_ASYNC
    // --- prologue: async-stage tile 0 into LDS[0]
    __builtin_amdgcn_global_load_async_to_lds_b128((as1_v4i*)pa0, (as3_v4i*)sA0, 0, 0);
    __builtin_amdgcn_global_load_async_to_lds_b128((as1_v4i*)pa1, (as3_v4i*)sA1, 0, 0);
    __builtin_amdgcn_global_load_async_to_lds_b128((as1_v4i*)pb0, (as3_v4i*)sB0, 0, 0);
    __builtin_amdgcn_global_load_async_to_lds_b128((as1_v4i*)pb1, (as3_v4i*)sB1, 0, 0);
    __builtin_amdgcn_s_wait_asynccnt(0);
    __syncthreads();
#else
    uint4 ra0 = *(const uint4*)(pa0);
    uint4 ra1 = *(const uint4*)(pa1);
    uint4 rb0 = *(const uint4*)(pb0);
    uint4 rb1 = *(const uint4*)(pb1);
    *(uint4*)(sA0) = ra0; *(uint4*)(sA1) = ra1;
    *(uint4*)(sB0) = rb0; *(uint4*)(sB1) = rb1;
    __syncthreads();
#endif

    for (int k0 = 0; k0 < K; k0 += 32) {
        const int  buf  = (k0 >> 5) & 1;
        const bool last = (k0 + 32 >= K);
        const int  alt  = (buf ^ 1) * bufStride;

#if USE_ASYNC
        if (!last) {   // async-stage next tile into LDS[alt] (no VGPR staging)
            __builtin_amdgcn_global_load_async_to_lds_b128(
                (as1_v4i*)(pa0 + k0 + 32), (as3_v4i*)(sA0 + alt), 0, 0);
            __builtin_amdgcn_global_load_async_to_lds_b128(
                (as1_v4i*)(pa1 + k0 + 32), (as3_v4i*)(sA1 + alt), 0, 0);
            __builtin_amdgcn_global_load_async_to_lds_b128(
                (as1_v4i*)(pb0 + k0 + 32), (as3_v4i*)(sB0 + alt), 0, 0);
            __builtin_amdgcn_global_load_async_to_lds_b128(
                (as1_v4i*)(pb1 + k0 + 32), (as3_v4i*)(sB1 + alt), 0, 0);
        }
#else
        if (!last) {
            ra0 = *(const uint4*)(pa0 + k0 + 32);
            ra1 = *(const uint4*)(pa1 + k0 + 32);
            rb0 = *(const uint4*)(pb0 + k0 + 32);
            rb1 = *(const uint4*)(pb1 + k0 + 32);
        }
#endif
        if (k0 + 64 < K) {                 // L2 prefetch, 2 tiles ahead
            __builtin_prefetch(pa0 + k0 + 64, 0, 1);
            __builtin_prefetch(pb0 + k0 + 64, 0, 1);
        }

        // --- fragments from LDS[buf] + 2x4 WMMAs
        const unsigned short* baseA = lsA[0] + buf * bufStride;
        const unsigned short* baseB = lsB[0] + buf * bufStride;
        FragBF aF[2], bF[4];
        #pragma unroll
        for (int i = 0; i < 2; ++i) {
            const unsigned short* pr = baseA + (wm * 32 + i * 16 + ln) * 40;
            aF[i].q[0] = *(const uint4*)(pr + 8 * halfId);       // k = 8h..
            aF[i].q[1] = *(const uint4*)(pr + 16 + 8 * halfId);  // k = 16+8h..
        }
        #pragma unroll
        for (int j = 0; j < 4; ++j) {
            const unsigned short* pr =
                baseB + (wn * 64 + j * 16 + ln) * 40;
            bF[j].q[0] = *(const uint4*)(pr + 8 * halfId);       // k = 8h..
            bF[j].q[1] = *(const uint4*)(pr + 16 + 8 * halfId);  // k = 16+8h..
        }
        #pragma unroll
        for (int i = 0; i < 2; ++i)
            #pragma unroll
            for (int j = 0; j < 4; ++j)
                acc[i][j] = __builtin_amdgcn_wmma_f32_16x16x32_bf16(
                    false, aF[i].v, false, bF[j].v, (short)0, acc[i][j],
                    false, false);

#if USE_ASYNC
        if (!last) {
            __builtin_amdgcn_s_wait_asynccnt(0);   // my 4 transfers landed
            __syncthreads();                        // everyone's landed
        }
#else
        if (!last) {
            __syncthreads();
            *(uint4*)(sA0 + alt) = ra0; *(uint4*)(sA1 + alt) = ra1;
            *(uint4*)(sB0 + alt) = rb0; *(uint4*)(sB1 + alt) = rb1;
            __syncthreads();
        }
#endif
    }

    // --- epilogue + store (C layout: VGPR r -> row r + 8*halfId, col = ln)
    #pragma unroll
    for (int i = 0; i < 2; ++i) {
        #pragma unroll
        for (int j = 0; j < 4; ++j) {
            int mbase = blockM + wm * 32 + i * 16;
            int col   = blockN + wn * 64 + j * 16 + ln;
            if (col >= N) continue;
            #pragma unroll
            for (int r = 0; r < 8; ++r) {
                int row = mbase + halfId * 8 + r;   // M % 128 == 0: no guard
                float v = acc[i][j][r];
                if constexpr (EPI == 2) {
                    v += residual[(size_t)row * ldc + col];
                }
                if constexpr (EPI == 1) {
                    ((unsigned short*)Cout)[(size_t)row * ldc + col] = f2bf(v);
                } else {
                    ((float*)Cout)[(size_t)row * ldc + col] = v;
                }
            }
        }
    }
}

// ---------------------------------------------------------------------------
// Depthwise conv1d (K=3, same padding) + SiLU.  xb = xz[:, 0:D].
// ---------------------------------------------------------------------------
__global__ void k_conv_silu(const float* __restrict__ xz,
                            const float* __restrict__ cw,
                            const float* __restrict__ cb,
                            float* __restrict__ xact,
                            unsigned short* __restrict__ xactb,
                            int Mtot, int L, int D, int Dtwo) {
    int idx = blockIdx.x * blockDim.x + threadIdx.x;
    if (idx >= Mtot * D) return;
    int m = idx / D;
    int d = idx - m * D;
    int l = m % L;
    float w0 = cw[d], w1 = cw[D + d], w2 = cw[2 * D + d];
    float mid   = xz[(size_t)m * Dtwo + d];
    float left  = (l > 0)     ? xz[(size_t)(m - 1) * Dtwo + d] : 0.f;
    float right = (l < L - 1) ? xz[(size_t)(m + 1) * Dtwo + d] : 0.f;
    float v = w0 * left + w1 * mid + w2 * right + cb[d];
    float s = v / (1.0f + __expf(-v));      // SiLU
    xact[idx]  = s;
    xactb[idx] = f2bf(s);
}

// ---------------------------------------------------------------------------
// Selective scan, 3-phase segmented.  delta/Bm/Cm live in one fused buffer
// with row stride ld (cols [0,D)=delta_pre, [D,2D)=Bm, [2D,3D)=Cm).
// Phase 1 applies bias+softplus to delta_pre, then local scan; writes running
// product p over delta slot and h_loc over Bm slot.
// ---------------------------------------------------------------------------
__global__ void k_scan1(float* __restrict__ dslot,       // in: pre, out: prod
                        float* __restrict__ bslot,       // in: Bm,  out: h_loc
                        const float* __restrict__ xact,
                        const float* __restrict__ Avec,
                        const float* __restrict__ bdt,
                        float* __restrict__ carryP,
                        float* __restrict__ carryH,
                        int D, int Bb, int L, int LSEG, int ld) {
    int tid = blockIdx.x * blockDim.x + threadIdx.x;
    int d  = tid & (D - 1);
    int t2 = tid / D;
    int b  = t2 % Bb;
    int s  = t2 / Bb;
    int CH = Bb * D;
    int ch = b * D + d;
    size_t m0 = (size_t)b * L + (size_t)s * LSEG;
    float a_d = Avec[d];
    float bia = bdt[d];
    float h = 0.f, p = 1.f;
    for (int i = 0; i < LSEG; ++i) {
        size_t m   = m0 + i;
        size_t off = m * ld + d;
        float pre = dslot[off] + bia;
        float dl  = (pre > 20.0f) ? pre : log1pf(__expf(pre));   // softplus
        float bm  = bslot[off];
        float xa  = xact[m * D + d];
        float a   = __expf(dl * a_d);
        float bx  = dl * bm * xa;
        h = a * h + bx;
        p = p * a;
        dslot[off] = p;
        bslot[off] = h;
    }
    carryP[(size_t)s * CH + ch] = p;
    carryH[(size_t)s * CH + ch] = h;
}

// Phase 2: sequential carry combine across segments, one thread per channel.
__global__ void k_scan2(const float* __restrict__ carryP,
                        const float* __restrict__ carryH,
                        float* __restrict__ Hin, int CH, int NSEG) {
    int ch = blockIdx.x * blockDim.x + threadIdx.x;
    if (ch >= CH) return;
    float H = 0.f;
    for (int s = 0; s < NSEG; ++s) {
        size_t idx = (size_t)s * CH + ch;
        Hin[idx] = H;
        H = carryP[idx] * H + carryH[idx];
    }
}

// Phase 3: fix-up + y = C*h + D-skip + SiLU gating -> bf16 gemm input.
__global__ void k_scan3(const float* __restrict__ pslot,  // prod
                        const float* __restrict__ hslot,  // h_loc
                        const float* __restrict__ cslot,  // Cm
                        const float* __restrict__ Hin,
                        const float* __restrict__ xact,
                        const float* __restrict__ xz,     // z at cols [D,2D)
                        const float* __restrict__ Dskip,
                        unsigned short* __restrict__ gatedb,
                        int D, int Bb, int L, int LSEG, int ld, int Dtwo) {
    int tid = blockIdx.x * blockDim.x + threadIdx.x;
    int d  = tid & (D - 1);
    int t2 = tid / D;
    int b  = t2 % Bb;
    int s  = t2 / Bb;
    int CH = Bb * D;
    int ch = b * D + d;
    size_t m0 = (size_t)b * L + (size_t)s * LSEG;
    float Hi  = Hin[(size_t)s * CH + ch];
    float dsk = Dskip[d];
    for (int i = 0; i < LSEG; ++i) {
        size_t m   = m0 + i;
        size_t off = m * ld + d;
        float h  = pslot[off] * Hi + hslot[off];
        float y  = cslot[off] * h;
        float os = y + xact[m * D + d] * dsk;
        float zv = xz[m * Dtwo + D + d];
        float g  = os * (zv / (1.0f + __expf(-zv)));
        gatedb[m * D + d] = f2bf(g);
    }
}

// ---------------------------------------------------------------------------
// Host-side launcher.
// ---------------------------------------------------------------------------
extern "C" void kernel_launch(void* const* d_in, const int* in_sizes, int n_in,
                              void* d_out, int out_size, void* d_ws, size_t ws_size,
                              hipStream_t stream) {
    (void)in_sizes; (void)n_in; (void)out_size; (void)ws_size;

    constexpr int Bb = 2, L = 4096, D = 1024, DS = 16, RK = 64;
    constexpr int M    = Bb * L;     // 8192
    constexpr int Dtwo = 2 * D;      // 2048
    constexpr int NSEG = 64, LSEG = L / NSEG, CH = Bb * D;
    constexpr int KP2  = 128;        // fused proj width: [xp|Bg|0|Cg|0]
    constexpr int N3   = 3 * D;      // fused expand width: [dpre|Bm|Cm]

    const float* x      = (const float*)d_in[0];
    const float* gamma  = (const float*)d_in[1];
    const float* beta   = (const float*)d_in[2];
    const float* W_in   = (const float*)d_in[3];
    const float* conv_w = (const float*)d_in[4];
    const float* conv_b = (const float*)d_in[5];
    const float* W_xp   = (const float*)d_in[6];
    const float* W_Bg   = (const float*)d_in[7];
    const float* W_Cg   = (const float*)d_in[8];
    const float* W_dt   = (const float*)d_in[9];
    const float* b_dt   = (const float*)d_in[10];
    const float* W_Bp   = (const float*)d_in[11];
    const float* W_Cp   = (const float*)d_in[12];
    const float* Avec   = (const float*)d_in[13];
    const float* D_skip = (const float*)d_in[14];
    const float* W_out  = (const float*)d_in[15];
    float* out = (float*)d_out;

    // ---- workspace layout ----
    char* ws = (char*)d_ws;
    size_t off = 0;
    auto alloc = [&](size_t bytes) -> char* {
        char* p = ws + off;
        off = (off + bytes + 255) & ~(size_t)255;
        return p;
    };
    unsigned short* xnb    = (unsigned short*)alloc((size_t)M * D * 2);
    unsigned short* WinT   = (unsigned short*)alloc((size_t)Dtwo * D * 2);
    unsigned short* WoutT  = (unsigned short*)alloc((size_t)D * D * 2);
    unsigned short* BtP2   = (unsigned short*)alloc((size_t)KP2 * D * 2);  // [128][1024]
    unsigned short* BtP3   = (unsigned short*)alloc((size_t)N3 * KP2 * 2); // [3072][128]
    float*          xz     = (float*)alloc((size_t)M * Dtwo * 4);
    float*          xact   = (float*)alloc((size_t)M * D * 4);
    unsigned short* xactb  = (unsigned short*)alloc((size_t)M * D * 2);
    unsigned short* P2out  = (unsigned short*)alloc((size_t)M * KP2 * 2);  // [xp|Bg|0|Cg|0]
    float*          P3out  = (float*)alloc((size_t)M * N3 * 4);            // [dpre|Bm|Cm]
    float*          carryP = (float*)alloc((size_t)NSEG * CH * 4);
    float*          carryH = (float*)alloc((size_t)NSEG * CH * 4);
    float*          Hin    = (float*)alloc((size_t)NSEG * CH * 4);
    unsigned short* gatedb = (unsigned short*)alloc((size_t)M * D * 2);

    auto tgrid = [](int n) { return dim3((n + 255) / 256); };
    auto ggrid = [](int m, int n) { return dim3(m / 128, (n + 127) / 128); };

    // ---- 1. weight packing: fp32 [K][N] -> transposed bf16 blocks ----
    k_place_bf16<<<tgrid(Dtwo * D), 256, 0, stream>>>(W_in,  WinT,  D, Dtwo, D, 0, 0);
    k_place_bf16<<<tgrid(D * D),    256, 0, stream>>>(W_out, WoutT, D, D,    D, 0, 0);
    // BtP2 [128 rows][K=1024]: rows 0..63 Wxp^T, 64..79 WBg^T, 96..111 WCg^T
    k_zero_u16<<<tgrid(KP2 * D), 256, 0, stream>>>(BtP2, KP2 * D);
    k_place_bf16<<<tgrid(D * RK), 256, 0, stream>>>(W_xp, BtP2, D, RK, D, 0, 0);
    k_place_bf16<<<tgrid(D * DS), 256, 0, stream>>>(W_Bg, BtP2, D, DS, D, 0, 64);
    k_place_bf16<<<tgrid(D * DS), 256, 0, stream>>>(W_Cg, BtP2, D, DS, D, 0, 96);
    // BtP3 [3072 rows][K=128]: block-diagonal [W_dt^T @k0..63 | W_Bp^T @k64..79
    //                                          | W_Cp^T @k96..111]
    k_zero_u16<<<tgrid(N3 * KP2), 256, 0, stream>>>(BtP3, N3 * KP2);
    k_place_bf16<<<tgrid(RK * D), 256, 0, stream>>>(W_dt, BtP3, RK, D, KP2, 0,  0);
    k_place_bf16<<<tgrid(DS * D), 256, 0, stream>>>(W_Bp, BtP3, DS, D, KP2, 64, D);
    k_place_bf16<<<tgrid(DS * D), 256, 0, stream>>>(W_Cp, BtP3, DS, D, KP2, 96, 2 * D);

    // ---- 2. LayerNorm -> bf16 ----
    k_layernorm<<<M, 256, 0, stream>>>(x, gamma, beta, xnb, D);

    // ---- 3. in_proj GEMM: xz = xn @ W_in  (f32) ----
    k_gemm_wmma<0><<<ggrid(M, Dtwo), 256, 0, stream>>>(
        xnb, WinT, xz, nullptr, M, Dtwo, D, D, Dtwo);

    // ---- 4. depthwise conv + SiLU ----
    k_conv_silu<<<tgrid(M * D), 256, 0, stream>>>(xz, conv_w, conv_b,
                                                  xact, xactb, M, L, D, Dtwo);

    // ---- 5. fused low-rank gather GEMM: [xp|Bg|0|Cg|0] bf16 ----
    k_gemm_wmma<1><<<ggrid(M, KP2), 256, 0, stream>>>(
        xactb, BtP2, P2out, nullptr, M, KP2, D, D, KP2);

    // ---- 6. fused expand GEMM: [dpre|Bm|Cm] f32 ----
    k_gemm_wmma<0><<<ggrid(M, N3), 256, 0, stream>>>(
        P2out, BtP3, P3out, nullptr, M, N3, KP2, KP2, N3);

    // ---- 7. segmented selective scan + fused gating epilogue ----
    k_scan1<<<(CH * NSEG) / 256, 256, 0, stream>>>(
        P3out, P3out + D, xact, Avec, b_dt, carryP, carryH, D, Bb, L, LSEG, N3);
    k_scan2<<<(CH + 255) / 256, 256, 0, stream>>>(carryP, carryH, Hin, CH, NSEG);
    k_scan3<<<(CH * NSEG) / 256, 256, 0, stream>>>(
        P3out, P3out + D, P3out + 2 * D, Hin, xact, xz, D_skip, gatedb,
        D, Bb, L, LSEG, N3, Dtwo);

    // ---- 8. out_proj GEMM + residual ----
    k_gemm_wmma<2><<<ggrid(M, D), 256, 0, stream>>>(
        gatedb, WoutT, out, x, M, D, D, D, D);
}
